// Modelsubclass_91207925498496
// MI455X (gfx1250) — matrix operand order repaired and checked by reading, not compiled, not run on hardware
//
#include <hip/hip_runtime.h>

typedef __bf16 bf16;
typedef __attribute__((ext_vector_type(2)))  __bf16 bf16x2;
typedef __attribute__((ext_vector_type(8)))  __bf16 bf16x8;
typedef __attribute__((ext_vector_type(16))) __bf16 bf16x16;
typedef __attribute__((ext_vector_type(8)))  float  f32x8;
typedef __attribute__((ext_vector_type(4)))  int    i32x4;

#define DIM    512
#define SEQ    4096
#define BATCH  8
#define NTOK   (BATCH * SEQ)   // 32768
#define HEADS  8
#define DH     64
#define BLK    64
#define NBLK   64
#define LDV    80              // V^T tile stride (160B rows: 32B aligned)
#define LDP    72              // attn-prob tile stride (144B rows: 16B aligned)

#if __has_builtin(__builtin_amdgcn_global_load_async_to_lds_b128) && \
    __has_builtin(__builtin_amdgcn_s_wait_asynccnt)
#define HAS_ASYNC_LDS 1
#else
#define HAS_ASYNC_LDS 0
#endif

// 16-byte global -> LDS copy: async on gfx1250 (ASYNCcnt), sync fallback.
// Builtin signature: (v4i addrspace(1)* src, v4i addrspace(3)* dst,
//                     imm i32 offset, imm i32 cpol)
__device__ __forceinline__ void cp16(bf16* dst_lds, const bf16* src_glob) {
#if HAS_ASYNC_LDS
  __builtin_amdgcn_global_load_async_to_lds_b128(
      (__attribute__((address_space(1))) i32x4*)src_glob,
      (__attribute__((address_space(3))) i32x4*)dst_lds, 0, 0);
#else
  *(uint4*)dst_lds = *(const uint4*)src_glob;
#endif
}

__device__ __forceinline__ void wait_async_lds() {
#if HAS_ASYNC_LDS
  __builtin_amdgcn_s_wait_asynccnt(0);
#endif
}

// ---------------------------------------------------------------------------
// WMMA fragment loaders (wave32, v_wmma_f32_16x16x32_bf16).
// A (16x32, [M][K] row-major): lane&15 = M; lanes 0-15 hold K 0-15,
// lanes 16-31 hold K 16-31 (same k order as B).
// B (32x16): lane&15 = N; lanes 0-15 hold K 0-15, lanes 16-31 hold K 16-31;
// loaded from [N][K] row-major storage -> one contiguous 32B vector load.
// ---------------------------------------------------------------------------
__device__ __forceinline__ bf16x16 frag_a(const bf16* p, int ld, int lane) {
  const bf16* r = p + (size_t)(lane & 15) * ld + ((lane >> 4) << 4);
  bf16x8 lo = *(const bf16x8*)r;
  bf16x8 hi = *(const bf16x8*)(r + 8);
  return __builtin_shufflevector(lo, hi, 0, 1, 2, 3, 4, 5, 6, 7,
                                 8, 9, 10, 11, 12, 13, 14, 15);
}

__device__ __forceinline__ bf16x16 frag_b_nk(const bf16* p, int ld, int lane) {
  return *(const bf16x16*)(p + (size_t)(lane & 15) * ld + ((lane >> 4) << 4));
}

__device__ __forceinline__ f32x8 wmma_bf16(bf16x16 a, bf16x16 b, f32x8 c) {
  return __builtin_amdgcn_wmma_f32_16x16x32_bf16(false, a, false, b,
                                                 (short)0, c, false, false);
}

// ---------------------------------------------------------------------------
// Kernel 1: x = (emb + pos_enc) * mask  ->  fp32 (residual) + bf16 (WMMA A)
// ---------------------------------------------------------------------------
__global__ __launch_bounds__(128)
void prep_kernel(const float* __restrict__ emb, const int* __restrict__ ids,
                 float* __restrict__ xf, bf16* __restrict__ xh) {
  const int tok = blockIdx.x;
  const int s = tok & (SEQ - 1);
  const float m = (ids[tok] != 0) ? 1.0f : 0.0f;
  const int d0 = threadIdx.x * 4;
#pragma unroll
  for (int j = 0; j < 4; ++j) {
    const int d = d0 + j;
    // 10000^(-(2*(d/2))/DIM) = exp(-(d & ~1) * ln(1e4)/DIM)
    const float freq = __expf(-(float)(d & ~1) * (9.2103403719762f / (float)DIM));
    const float ang = (float)s * freq;
    const float pe = (d & 1) ? __cosf(ang) : __sinf(ang);
    const float v = (emb[(size_t)tok * DIM + d] + pe) * m;
    xf[(size_t)tok * DIM + d] = v;
    xh[(size_t)tok * DIM + d] = (bf16)v;
  }
}

// ---------------------------------------------------------------------------
// Kernel 2: w1/w2/w3 fp32 [K][N] -> bf16 TRANSPOSED [3][N][K], so the GEMM's
// B tiles are contiguous-per-column and fragments are single vector loads.
// ---------------------------------------------------------------------------
__global__ __launch_bounds__(256)
void wcvt_kernel(const float* __restrict__ w1, const float* __restrict__ w2,
                 const float* __restrict__ w3, bf16* __restrict__ out) {
  const int i = blockIdx.x * 256 + threadIdx.x;  // over DIM*DIM
  const int n = i >> 9, k = i & (DIM - 1);
  const float* src = (blockIdx.y == 0) ? w1 : (blockIdx.y == 1) ? w2 : w3;
  out[(size_t)blockIdx.y * DIM * DIM + (size_t)n * DIM + k] =
      (bf16)src[(size_t)k * DIM + n];
}

// ---------------------------------------------------------------------------
// Kernel 3: QKV GEMM.  C[t][n] = sum_k x[t][k] W[k][n] + b[n]  (bf16 out)
// grid (M/128, N/64, 3); 8 waves; double-buffered async global->LDS pipeline.
// Q (z==0) folds the 1/sqrt(DH)=0.125 attention scale.
// ---------------------------------------------------------------------------
__global__ __launch_bounds__(256)
void qkv_gemm_kernel(const bf16* __restrict__ X, const bf16* __restrict__ Wt,
                     const float* __restrict__ b1, const float* __restrict__ b2,
                     const float* __restrict__ b3, bf16* __restrict__ QKV) {
  const int z = blockIdx.z;
  const bf16* W = Wt + (size_t)z * DIM * DIM;          // [N][K] bf16
  const float* bias = (z == 0) ? b1 : (z == 1) ? b2 : b3;
  bf16* out = QKV + (size_t)z * NTOK * DIM;
  const float scale = (z == 0) ? 0.125f : 1.0f;

  const int m0 = blockIdx.x * 128;
  const int n0 = blockIdx.y * 64;
  const int t = threadIdx.x;
  const int wave = t >> 5, lane = t & 31;

  __shared__ __align__(32) bf16 As[2][128 * 32];       // [M][K] tiles
  __shared__ __align__(32) bf16 Bs[2][64 * 32];        // [N][K] tiles

  f32x8 acc[4] = {};

  auto load_chunk = [&](int kc, int buf) {
    // A: 512 16B units, 2 per thread
#pragma unroll
    for (int j = 0; j < 2; ++j) {
      const int u = t + 256 * j;
      const int row = u >> 2, seg = u & 3;
      const bf16* src = X + (size_t)(m0 + row) * DIM + kc * 32 + seg * 8;
      cp16(&As[buf][row * 32 + seg * 8], src);
      if (j == 0) __builtin_prefetch(src + 64, 0, 0);  // global_prefetch_b8
    }
    // B: 256 16B units, 1 per thread
    const int row = t >> 2, seg = t & 3;
    cp16(&Bs[buf][row * 32 + seg * 8],
         W + (size_t)(n0 + row) * DIM + kc * 32 + seg * 8);
  };

  load_chunk(0, 0);
#pragma unroll 2
  for (int kc = 0; kc < DIM / 32; ++kc) {
    wait_async_lds();
    __syncthreads();                                   // chunk kc visible
    if (kc + 1 < DIM / 32) load_chunk(kc + 1, (kc + 1) & 1);

    const bf16x16 a = frag_a(&As[kc & 1][(wave * 16) * 32], 32, lane);
#pragma unroll
    for (int ng = 0; ng < 4; ++ng) {
      const bf16x16 b = frag_b_nk(&Bs[kc & 1][(ng * 16) * 32], 32, lane);
      acc[ng] = wmma_bf16(a, b, acc[ng]);
    }
  }

  const int rbase = (lane >> 4) * 8;
  const int col = lane & 15;
#pragma unroll
  for (int ng = 0; ng < 4; ++ng) {
    const int gc = n0 + ng * 16 + col;
    const float bv = bias[gc];
#pragma unroll
    for (int i = 0; i < 8; ++i) {
      const int gr = m0 + wave * 16 + rbase + i;
      out[(size_t)gr * DIM + gc] = (bf16)((acc[ng][i] + bv) * scale);
    }
  }
}

// ---------------------------------------------------------------------------
// Kernel 4: block-local attention. One WG per (batch, block); one wave per
// head. scores = (Q/8) K^T (WMMA) -> softmax (register, shfl over 16-lane
// halves; the reference's uniform per-row mask add is a softmax no-op so it
// is dropped exactly) -> P V (WMMA, V staged transposed in LDS). fp32 out.
// ---------------------------------------------------------------------------
__global__ __launch_bounds__(256)
void attn_kernel(const bf16* __restrict__ Q, const bf16* __restrict__ K,
                 const bf16* __restrict__ V, float* __restrict__ O) {
  const int t0 = blockIdx.x * BLK;
  const int h = threadIdx.x >> 5;
  const int lane = threadIdx.x & 31;
  const int hd = h * DH;

  __shared__ __align__(32) bf16 Vt[HEADS][DH * LDV];  // [head][n(dh)][k(tok)]
  __shared__ __align__(16) bf16 P[HEADS][16 * LDP];   // probs, 16-row strip

  // stage V^T for this head: lane owns token pair (2*lane, 2*lane+1),
  // packs them into one b32 store per dh element.
  {
    const int k0 = lane * 2;
    const bf16* v0 = V + (size_t)(t0 + k0) * DIM + hd;
    const bf16* v1 = v0 + DIM;
#pragma unroll 8
    for (int n = 0; n < DH; ++n) {
      bf16x2 p2;
      p2[0] = v0[n];
      p2[1] = v1[n];
      *(bf16x2*)&Vt[h][n * LDV + k0] = p2;
    }
  }
  __syncthreads();

  for (int qg = 0; qg < 4; ++qg) {
    const bf16* qbase = Q + (size_t)(t0 + qg * 16) * DIM + hd;
    const bf16x16 aQ0 = frag_a(qbase, DIM, lane);
    const bf16x16 aQ1 = frag_a(qbase + 32, DIM, lane);

    f32x8 s[4] = {};
#pragma unroll
    for (int nt = 0; nt < 4; ++nt) {
      const bf16* kbase = K + (size_t)(t0 + nt * 16) * DIM + hd;
      s[nt] = wmma_bf16(aQ0, frag_b_nk(kbase, DIM, lane), s[nt]);
      s[nt] = wmma_bf16(aQ1, frag_b_nk(kbase + 32, DIM, lane), s[nt]);
    }

    // row softmax: rows live in a fixed 16-lane half; reduce via shfl_xor
#pragma unroll
    for (int i = 0; i < 8; ++i) {
      float rmax = -3.0e38f;
#pragma unroll
      for (int nt = 0; nt < 4; ++nt) {
        float v = s[nt][i];
        v = fmaxf(v, __shfl_xor(v, 1, 32));
        v = fmaxf(v, __shfl_xor(v, 2, 32));
        v = fmaxf(v, __shfl_xor(v, 4, 32));
        v = fmaxf(v, __shfl_xor(v, 8, 32));
        rmax = fmaxf(rmax, v);
      }
      float rsum = 0.0f;
#pragma unroll
      for (int nt = 0; nt < 4; ++nt) {
        float e = __expf(s[nt][i] - rmax);
        s[nt][i] = e;
        e += __shfl_xor(e, 1, 32);
        e += __shfl_xor(e, 2, 32);
        e += __shfl_xor(e, 4, 32);
        e += __shfl_xor(e, 8, 32);
        rsum += e;
      }
      const float rinv = __frcp_rn(rsum);
#pragma unroll
      for (int nt = 0; nt < 4; ++nt) s[nt][i] *= rinv;
    }

    // redistribute P from C-layout (lane = key col) to A-layout (lane = q row)
    const int prow = (lane >> 4) * 8;
    const int pcol = lane & 15;
#pragma unroll
    for (int nt = 0; nt < 4; ++nt)
#pragma unroll
      for (int i = 0; i < 8; ++i)
        P[h][(prow + i) * LDP + nt * 16 + pcol] = (bf16)s[nt][i];
    __syncthreads();

    const bf16x16 aP0 = frag_a(&P[h][0], LDP, lane);
    const bf16x16 aP1 = frag_a(&P[h][32], LDP, lane);

    f32x8 o[4] = {};
#pragma unroll
    for (int nt = 0; nt < 4; ++nt) {
      o[nt] = wmma_bf16(aP0, frag_b_nk(&Vt[h][(nt * 16) * LDV], LDV, lane), o[nt]);
      o[nt] = wmma_bf16(aP1, frag_b_nk(&Vt[h][(nt * 16) * LDV + 32], LDV, lane), o[nt]);
    }

#pragma unroll
    for (int nt = 0; nt < 4; ++nt)
#pragma unroll
      for (int i = 0; i < 8; ++i) {
        const int gr = t0 + qg * 16 + prow + i;
        O[(size_t)gr * DIM + hd + nt * 16 + pcol] = o[nt][i];
      }
    __syncthreads();
  }
}

// ---------------------------------------------------------------------------
// Kernel 5: residual + LayerNorm (eps 1e-3). One wave per token, float4 I/O.
// Row statistics are permutation-invariant so contiguous per-lane chunks OK.
// ---------------------------------------------------------------------------
__global__ __launch_bounds__(256)
void ln_kernel(const float* __restrict__ A, const float* __restrict__ X,
               float* __restrict__ out) {
  const int tok = blockIdx.x * 8 + (threadIdx.x >> 5);
  const int lane = threadIdx.x & 31;
  const float4* pa = (const float4*)(A + (size_t)tok * DIM + lane * 16);
  const float4* px = (const float4*)(X + (size_t)tok * DIM + lane * 16);
  float4 y[4];
  float s = 0.0f, sq = 0.0f;
#pragma unroll
  for (int j = 0; j < 4; ++j) {
    const float4 a = pa[j], b = px[j];
    const float4 v = make_float4(a.x + b.x, a.y + b.y, a.z + b.z, a.w + b.w);
    y[j] = v;
    s += v.x + v.y + v.z + v.w;
    sq += v.x * v.x + v.y * v.y + v.z * v.z + v.w * v.w;
  }
#pragma unroll
  for (int m = 16; m >= 1; m >>= 1) {
    s += __shfl_xor(s, m, 32);
    sq += __shfl_xor(sq, m, 32);
  }
  const float mean = s * (1.0f / DIM);
  const float var = sq * (1.0f / DIM) - mean * mean;
  const float rstd = __frsqrt_rn(fmaxf(var, 0.0f) + 1e-3f);
  float4* po = (float4*)(out + (size_t)tok * DIM + lane * 16);
#pragma unroll
  for (int j = 0; j < 4; ++j) {
    po[j] = make_float4((y[j].x - mean) * rstd, (y[j].y - mean) * rstd,
                        (y[j].z - mean) * rstd, (y[j].w - mean) * rstd);
  }
}

// ---------------------------------------------------------------------------
extern "C" void kernel_launch(void* const* d_in, const int* in_sizes, int n_in,
                              void* d_out, int out_size, void* d_ws, size_t ws_size,
                              hipStream_t stream) {
  const float* emb = (const float*)d_in[0];
  const float* w1 = (const float*)d_in[1];
  const float* b1 = (const float*)d_in[2];
  const float* w2 = (const float*)d_in[3];
  const float* b2 = (const float*)d_in[4];
  const float* w3 = (const float*)d_in[5];
  const float* b3 = (const float*)d_in[6];
  const int* ids = (const int*)d_in[7];

  char* ws = (char*)d_ws;
  const size_t xf_bytes = (size_t)NTOK * DIM * 4;       // 64 MiB fp32 x
  const size_t xh_bytes = (size_t)NTOK * DIM * 2;       // 32 MiB bf16 x
  const size_t wb_bytes = (size_t)3 * DIM * DIM * 2;    // 1.5 MiB bf16 W^T
  const size_t qkv_bytes = (size_t)3 * NTOK * DIM * 2;  // 96 MiB bf16 QKV

  float* xf = (float*)ws;
  bf16* xh = (bf16*)(ws + xf_bytes);
  bf16* wb = (bf16*)(ws + xf_bytes + xh_bytes);
  bf16* qkv = (bf16*)(ws + xf_bytes + xh_bytes + wb_bytes);
  float* attn = (float*)(ws + xf_bytes + xh_bytes + wb_bytes + qkv_bytes);

  prep_kernel<<<NTOK, 128, 0, stream>>>(emb, ids, xf, xh);
  wcvt_kernel<<<dim3(DIM * DIM / 256, 3), 256, 0, stream>>>(w1, w2, w3, wb);
  qkv_gemm_kernel<<<dim3(NTOK / 128, DIM / 64, 3), 256, 0, stream>>>(
      xh, wb, b1, b2, b3, qkv);
  attn_kernel<<<BATCH * NBLK, 256, 0, stream>>>(
      qkv, qkv + (size_t)NTOK * DIM, qkv + (size_t)2 * NTOK * DIM, attn);
  ln_kernel<<<NTOK / 8, 256, 0, stream>>>(attn, xf, (float*)d_out);
}
